// NTXentLoss_20177756357053
// MI455X (gfx1250) — compile-verified, matrix operand order checked by reading, not run on hardware
//
#include <hip/hip_runtime.h>
#include <hip/hip_bf16.h>
#include <cstddef>

// ---------------------------------------------------------------------------
// NT-Xent loss with hard-negative up-weighting, MI455X (gfx1250, wave32).
//   B = 8192, D = 1024, temperature 0.05, hard-negative weight 0.5.
//
// Fully fused plan (no 256MB sim materialization):
//   1) bf16x3-split WMMA GEMM, 128x128 tiles. Each block owns the complete
//      K-reduction for its tile, so the epilogue reduces the tile into
//      per-row and per-column partial softmax/argmax states -> 20MB workspace.
//   2) Deterministic chunk merges (ascending index order) -> per-row losses.
//   3) Scalar reduction -> d_out[0].
// ---------------------------------------------------------------------------

#define BDIM 8192
#define DDIM 1024
#define NCHUNK (BDIM / 128)        // 64 partial chunks per row / per column

typedef __attribute__((ext_vector_type(16))) __bf16 bf16x16;
typedef __attribute__((ext_vector_type(8)))  float  f32x8;

constexpr int BM = 128;            // block tile M
constexpr int BN = 128;            // block tile N
constexpr int BK = 32;             // K step (one bf16 WMMA depth)
constexpr int LSTR = BK + 8;       // LDS row stride (40 bf16 = 80B = 20 banks)
constexpr float INV_T = 1.0f / 0.05f;   // == 20.0f exactly in fp32 (1-ulp vs /0.05f)

// ------------------- online softmax / masked-argmax state ------------------
struct St { float M, S, hv, d; int hi; };

__device__ __forceinline__ St st_init() {
    St s; s.M = -3.0e38f; s.S = 0.0f; s.hv = -3.0e38f; s.d = 0.0f; s.hi = 0x7fffffff;
    return s;
}

__device__ __forceinline__ void st_add(St& s, float x, int idx, bool isdiag) {
    const float mv = isdiag ? 0.0f : x;          // reference masks diag to 0
    if (mv > s.hv || (mv == s.hv && idx < s.hi)) { s.hv = mv; s.hi = idx; }
    if (isdiag) s.d = x;
    if (x > s.M) { s.S = s.S * __expf(s.M - x) + 1.0f; s.M = x; }
    else         { s.S += __expf(x - s.M); }
}

__device__ __forceinline__ St st_merge(const St& a, const St& b) {
    St r;
    const float Mn = fmaxf(a.M, b.M);
    r.S = a.S * __expf(a.M - Mn) + b.S * __expf(b.M - Mn);
    r.M = Mn;
    if (b.hv > a.hv || (b.hv == a.hv && b.hi < a.hi)) { r.hv = b.hv; r.hi = b.hi; }
    else                                              { r.hv = a.hv; r.hi = a.hi; }
    r.d = a.d + b.d;                 // exactly one contributor holds the diagonal
    return r;
}

__device__ __forceinline__ St st_shfl_xor(const St& s, int o) {
    St r;
    r.M  = __shfl_xor(s.M, o);
    r.S  = __shfl_xor(s.S, o);
    r.hv = __shfl_xor(s.hv, o);
    r.d  = __shfl_xor(s.d, o);
    r.hi = __shfl_xor(s.hi, o);
    return r;
}

// final: patch the hard-negative logit (x1.5), stable LSE, -log p[diag]
__device__ __forceinline__ float st_loss(const St& s, int r) {
    const float raw_h = (s.hi == r) ? s.d : s.hv;   // masked-diag degenerate case
    const float h2    = 1.5f * raw_h;
    const float Mn    = fmaxf(s.M, h2);
    const float S2    = s.S * __expf(s.M - Mn) - __expf(raw_h - Mn) + __expf(h2 - Mn);
    const float diag  = (s.hi == r) ? h2 : s.d;
    return (Mn + __logf(S2)) - diag;
}

// --------------------------- fused GEMM + reduce ---------------------------
__global__ __launch_bounds__(256)
void sim_gemm_fused(const float* __restrict__ A,   // [B, D] image embeddings
                    const float* __restrict__ Bm,  // [B, D] song embeddings
                    float* __restrict__ rM, float* __restrict__ rS,
                    float* __restrict__ rHV, float* __restrict__ rD,
                    int* __restrict__ rHI,
                    float* __restrict__ cM, float* __restrict__ cS,
                    float* __restrict__ cHV, float* __restrict__ cD,
                    int* __restrict__ cHI)
{
    __shared__ __bf16 Ah[BM * LSTR];
    __shared__ __bf16 Al[BM * LSTR];
    __shared__ __bf16 Bh[BN * LSTR];
    __shared__ __bf16 Bl[BN * LSTR];
    // partial-state staging: rows [128][2 waveN], cols [128][4 waveM]
    __shared__ float RPf[4][128 * 2];  __shared__ int RPi[128 * 2];
    __shared__ float CPf[4][128 * 4];  __shared__ int CPi[128 * 4];

    const int tid   = threadIdx.x;
    const int lane  = tid & 31;
    const int wid   = tid >> 5;            // 8 waves
    const int waveM = wid >> 1;            // 0..3  -> 32-row strip
    const int waveN = wid & 1;             // 0..1  -> 64-col strip
    const int bm0   = blockIdx.x * BM;
    const int bn0   = blockIdx.y * BN;
    const int half  = lane >> 4;
    const int n16   = lane & 15;

    f32x8 acc[2][4];
    #pragma unroll
    for (int mt = 0; mt < 2; ++mt)
        #pragma unroll
        for (int nt = 0; nt < 4; ++nt)
            #pragma unroll
            for (int e = 0; e < 8; ++e)
                acc[mt][nt][e] = 0.0f;

    // global-tile coordinates for this thread's 4 float4 loads
    const int fi_row[4] = { (0*256 + tid) >> 3, (1*256 + tid) >> 3,
                            (2*256 + tid) >> 3, (3*256 + tid) >> 3 };
    const int fi_c4[4]  = { (0*256 + tid) & 7, (1*256 + tid) & 7,
                            (2*256 + tid) & 7, (3*256 + tid) & 7 };

    float4 pa[4], pb[4];
    // prefetch K-tile 0
    #pragma unroll
    for (int j = 0; j < 4; ++j) {
        pa[j] = *reinterpret_cast<const float4*>(
            &A[(size_t)(bm0 + fi_row[j]) * DDIM + fi_c4[j] * 4]);
        pb[j] = *reinterpret_cast<const float4*>(
            &Bm[(size_t)(bn0 + fi_row[j]) * DDIM + fi_c4[j] * 4]);
    }

    for (int k0 = 0; k0 < DDIM; k0 += BK) {
        // ---- store the prefetched tile to LDS with bf16 hi/lo split ----
        #pragma unroll
        for (int j = 0; j < 4; ++j) {
            const float af[4] = {pa[j].x, pa[j].y, pa[j].z, pa[j].w};
            const float bf[4] = {pb[j].x, pb[j].y, pb[j].z, pb[j].w};
            #pragma unroll
            for (int c = 0; c < 4; ++c) {
                const int o = fi_row[j] * LSTR + fi_c4[j] * 4 + c;
                __bf16 h = (__bf16)af[c];
                Ah[o] = h;  Al[o] = (__bf16)(af[c] - (float)h);
                __bf16 g = (__bf16)bf[c];
                Bh[o] = g;  Bl[o] = (__bf16)(bf[c] - (float)g);
            }
        }
        __syncthreads();

        // ---- issue next K-tile's global loads (overlap with WMMA below) ----
        if (k0 + BK < DDIM) {
            #pragma unroll
            for (int j = 0; j < 4; ++j) {
                pa[j] = *reinterpret_cast<const float4*>(
                    &A[(size_t)(bm0 + fi_row[j]) * DDIM + (k0 + BK) + fi_c4[j] * 4]);
                pb[j] = *reinterpret_cast<const float4*>(
                    &Bm[(size_t)(bn0 + fi_row[j]) * DDIM + (k0 + BK) + fi_c4[j] * 4]);
            }
        }

        // ---- load WMMA fragments (16-bit A/B layout: lane%16 = M (resp N),
        //      K pairs split at lane 16 per the CDNA5 ISA striping) ----
        bf16x16 aH[2], aL[2], bH[4], bL[4];
        #pragma unroll
        for (int mt = 0; mt < 2; ++mt) {
            const int r = waveM * 32 + mt * 16 + n16;
            #pragma unroll
            for (int v = 0; v < 8; ++v) {
                const int k = (v < 4) ? (half * 8 + 2 * v)
                                      : (16 + half * 8 + 2 * (v - 4));
                aH[mt][2*v]   = Ah[r * LSTR + k];
                aH[mt][2*v+1] = Ah[r * LSTR + k + 1];
                aL[mt][2*v]   = Al[r * LSTR + k];
                aL[mt][2*v+1] = Al[r * LSTR + k + 1];
            }
        }
        #pragma unroll
        for (int nt = 0; nt < 4; ++nt) {
            const int r = waveN * 64 + nt * 16 + n16;
            #pragma unroll
            for (int v = 0; v < 8; ++v) {
                const int k = (v < 4) ? (half * 8 + 2 * v)
                                      : (16 + half * 8 + 2 * (v - 4));
                bH[nt][2*v]   = Bh[r * LSTR + k];
                bH[nt][2*v+1] = Bh[r * LSTR + k + 1];
                bL[nt][2*v]   = Bl[r * LSTR + k];
                bL[nt][2*v+1] = Bl[r * LSTR + k + 1];
            }
        }

        // ---- bf16x3 accumulate: hi*hi + hi*lo + lo*hi ----
        #pragma unroll
        for (int mt = 0; mt < 2; ++mt)
            #pragma unroll
            for (int nt = 0; nt < 4; ++nt) {
                acc[mt][nt] = __builtin_amdgcn_wmma_f32_16x16x32_bf16(
                    false, aH[mt], false, bH[nt], (short)0, acc[mt][nt], false, false);
                acc[mt][nt] = __builtin_amdgcn_wmma_f32_16x16x32_bf16(
                    false, aH[mt], false, bL[nt], (short)0, acc[mt][nt], false, false);
                acc[mt][nt] = __builtin_amdgcn_wmma_f32_16x16x32_bf16(
                    false, aL[mt], false, bH[nt], (short)0, acc[mt][nt], false, false);
            }
        __syncthreads();   // protect LDS tiles before next iteration's stores
    }

    // =================== epilogue: in-tile partial reductions ===============
    // C/D fragment layout: element (v, lane) -> m = v + 8*half, n = lane&15.

    // ---- per-row states over this block's 128 columns ----
    #pragma unroll
    for (int mt = 0; mt < 2; ++mt)
        #pragma unroll
        for (int v = 0; v < 8; ++v) {
            const int row_l = waveM * 32 + mt * 16 + v + 8 * half;
            const int grow  = bm0 + row_l;
            St st = st_init();
            #pragma unroll
            for (int nt = 0; nt < 4; ++nt) {
                const int gcol = bn0 + waveN * 64 + nt * 16 + n16;
                st_add(st, acc[mt][nt][v] * INV_T, gcol, grow == gcol);
            }
            #pragma unroll
            for (int o = 1; o <= 8; o <<= 1)       // merge 16 lanes of a half
                st = st_merge(st, st_shfl_xor(st, o));
            if (n16 == 0) {
                const int s = row_l * 2 + waveN;
                RPf[0][s] = st.M; RPf[1][s] = st.S; RPf[2][s] = st.hv;
                RPf[3][s] = st.d; RPi[s] = st.hi;
            }
        }

    // ---- per-column states over this block's 128 rows ----
    #pragma unroll
    for (int nt = 0; nt < 4; ++nt) {
        const int col_l = waveN * 64 + nt * 16 + n16;
        const int gcol  = bn0 + col_l;
        St st = st_init();
        #pragma unroll
        for (int mt = 0; mt < 2; ++mt)
            #pragma unroll
            for (int v = 0; v < 8; ++v) {
                const int grow = bm0 + waveM * 32 + mt * 16 + v + 8 * half;
                st_add(st, acc[mt][nt][v] * INV_T, grow, grow == gcol);
            }
        st = st_merge(st, st_shfl_xor(st, 16));    // join the two halves
        if (half == 0) {
            const int s = col_l * 4 + waveM;
            CPf[0][s] = st.M; CPf[1][s] = st.S; CPf[2][s] = st.hv;
            CPf[3][s] = st.d; CPi[s] = st.hi;
        }
    }
    __syncthreads();

    // ---- fold cross-wave parts (ascending index order) and store chunks ----
    if (tid < 128) {
        St st = st_init();
        #pragma unroll
        for (int w = 0; w < 2; ++w) {
            St p; const int s = tid * 2 + w;
            p.M = RPf[0][s]; p.S = RPf[1][s]; p.hv = RPf[2][s];
            p.d = RPf[3][s]; p.hi = RPi[s];
            st = st_merge(st, p);
        }
        const size_t o = (size_t)blockIdx.y * BDIM + (bm0 + tid);
        rM[o] = st.M; rS[o] = st.S; rHV[o] = st.hv; rD[o] = st.d; rHI[o] = st.hi;
    } else if (tid < 256) {
        const int t = tid - 128;
        St st = st_init();
        #pragma unroll
        for (int w = 0; w < 4; ++w) {
            St p; const int s = t * 4 + w;
            p.M = CPf[0][s]; p.S = CPf[1][s]; p.hv = CPf[2][s];
            p.d = CPf[3][s]; p.hi = CPi[s];
            st = st_merge(st, p);
        }
        const size_t o = (size_t)blockIdx.x * BDIM + (bn0 + t);
        cM[o] = st.M; cS[o] = st.S; cHV[o] = st.hv; cD[o] = st.d; cHI[o] = st.hi;
    }
}

// ----------------- merge 64 chunks per row/column -> loss ------------------
__global__ __launch_bounds__(256)
void merge_loss(const float* __restrict__ M, const float* __restrict__ S,
                const float* __restrict__ HV, const float* __restrict__ D,
                const int* __restrict__ HI, float* __restrict__ loss)
{
    const int r = blockIdx.x * 256 + threadIdx.x;
    St st = st_init();
    #pragma unroll 4
    for (int ch = 0; ch < NCHUNK; ++ch) {         // ascending => argmax tie-safe
        const size_t o = (size_t)ch * BDIM + r;
        St p; p.M = M[o]; p.S = S[o]; p.hv = HV[o]; p.d = D[o]; p.hi = HI[o];
        st = st_merge(st, p);
    }
    loss[r] = st_loss(st, r);
}

// ------------------------------ final reduce -------------------------------
__global__ __launch_bounds__(256)
void final_reduce(const float* __restrict__ losses,  // [2*B]
                  float* __restrict__ out)
{
    __shared__ float sm[256];
    float s = 0.0f;
    for (int i = threadIdx.x; i < 2 * BDIM; i += 256) s += losses[i];
    sm[threadIdx.x] = s;
    __syncthreads();
    for (int o = 128; o > 0; o >>= 1) {
        if (threadIdx.x < o) sm[threadIdx.x] += sm[threadIdx.x + o];
        __syncthreads();
    }
    if (threadIdx.x == 0)
        out[0] = sm[0] / (2.0f * (float)BDIM);   // (mean_i2s + mean_s2i)/2
}

// ------------------------------- launcher ----------------------------------
extern "C" void kernel_launch(void* const* d_in, const int* in_sizes, int n_in,
                              void* d_out, int out_size, void* d_ws, size_t ws_size,
                              hipStream_t stream)
{
    (void)in_sizes; (void)n_in; (void)out_size; (void)ws_size;
    const float* img = (const float*)d_in[0];
    const float* sng = (const float*)d_in[1];

    const size_t CH = (size_t)NCHUNK * BDIM;     // 64*8192 elements per field
    float* p = (float*)d_ws;
    float* rM = p;            float* rS = p + CH;    float* rHV = p + 2*CH;
    float* rD = p + 3*CH;     int*   rHI = (int*)(p + 4*CH);
    float* q  = p + 5*CH;
    float* cM = q;            float* cS = q + CH;    float* cHV = q + 2*CH;
    float* cD = q + 3*CH;     int*   cHI = (int*)(q + 4*CH);
    float* losses = q + 5*CH;                    // 2*B floats

    dim3 ggrid(BDIM / BM, BDIM / BN);
    sim_gemm_fused<<<ggrid, dim3(256), 0, stream>>>(img, sng,
        rM, rS, rHV, rD, rHI, cM, cS, cHV, cD, cHI);
    merge_loss<<<BDIM / 256, 256, 0, stream>>>(rM, rS, rHV, rD, rHI, losses);
    merge_loss<<<BDIM / 256, 256, 0, stream>>>(cM, cS, cHV, cD, cHI, losses + BDIM);
    final_reduce<<<1, 256, 0, stream>>>(losses, (float*)d_out);
}
